// NMRBackbone_85976655331861
// MI455X (gfx1250) — compile-verified, matrix operand order checked by reading, not run on hardware
//
#include <hip/hip_runtime.h>
#include <hip/hip_bf16.h>

// ---------------------------------------------------------------------------
// Types for CDNA5 WMMA (wave32): V_WMMA_F32_16X16X32_BF16
// ---------------------------------------------------------------------------
typedef __attribute__((ext_vector_type(16))) __bf16 v16bf;
typedef __attribute__((ext_vector_type(8)))  __bf16 v8bf;
typedef __attribute__((ext_vector_type(8)))  float  v8f;

#define WAVES_PER_BLOCK 8
#define NT 8   // 8 x 16 = 128-wide output strip per wave

// ---------------------------------------------------------------------------
// Activations
// ---------------------------------------------------------------------------
__device__ __forceinline__ float act_ssp(float v) {
  float a = fmaxf(v, 0.f);
  return a + log1pf(__expf(-fabsf(v))) - 0.69314718055994531f;
}
__device__ __forceinline__ float act_silu(float v) {
  return v * (1.f / (1.f + __expf(-v)));
}

// ---------------------------------------------------------------------------
// Weight repack: f32 [K,N] row-major  ->  bf16 WMMA-B fragment layout.
// Fragment order: frag(kc, jt) at offset ((kc*(N/16)+jt)*32 + lane)*16,
// each lane's 16 elements contiguous (one 32B load at GEMM time).
// Element map (ISA 7.12.2 B 32x16 bf16): col = jt*16 + (lane&15),
// koff = (lane>>4)*8;  e<8 -> K=kc*32+koff+e ; e>=8 -> K=kc*32+koff+16+(e-8).
// ---------------------------------------------------------------------------
__global__ void k_repack_b(const float* __restrict__ B, __bf16* __restrict__ Bp,
                           int K, int N) {
  int idx = blockIdx.x * blockDim.x + threadIdx.x;
  int total = (K >> 5) * (N >> 4) * 32;
  if (idx >= total) return;
  int lane = idx & 31;
  int rest = idx >> 5;
  int jt = rest % (N >> 4);
  int kc = rest / (N >> 4);
  int col = jt * 16 + (lane & 15);
  int kb = kc * 32 + ((lane >> 4) << 3);
  v16bf v;
#pragma unroll
  for (int e = 0; e < 8; ++e) {
    v[e]     = (__bf16)B[(long)(kb + e) * N + col];
    v[8 + e] = (__bf16)B[(long)(kb + 16 + e) * N + col];
  }
  *(v16bf*)(Bp + (long)idx * 16) = v;
}

// ---------------------------------------------------------------------------
// Generic fused GEMM:  C[M,N] = epi( sum_i gather(A_i) @ B_i + bias + r1m*r1n )
// A_i bf16 [M,K_i] row-major (optional row gather); B_i repacked (above).
// Each wave: 16(M) x 128(N) strip, 8 accumulators share one A fragment.
// ---------------------------------------------------------------------------
struct GemmP {
  const __bf16* A0; const int* g0; int K0; const __bf16* B0;
  const __bf16* A1; const int* g1; int K1; const __bf16* B1;
  const __bf16* A2; const int* g2; int K2; const __bf16* B2;
  const float* bias;          // [N] or null
  const float* r1m;           // [M] or null (rank-1 term, e.g. d^2)
  const float* r1n;           // [N] or null (rank-1 weight row, f32)
  const float* mulB;          // [M,N] or null (post-act multiply, SchNet filt)
  const float* resB;          // [M,N] or null (residual add)
  const float* gA;            // [M,N] gate branch A (hs) for ACT==3
  const float* gB;            // [M,N] gate branch B (he) for ACT==3
  float*       Cf;            // f32 out (or scatter target) or null
  __bf16*      Cb;            // bf16 out or null
  const int*   scat;          // [M] scatter row (atomicAdd into Cf) or null
  int M; int N;
};

__device__ __forceinline__ void gemm_accum(const __bf16* __restrict__ A,
                                           const int* __restrict__ g, int K,
                                           const __bf16* __restrict__ Bp,
                                           int rowm, int lane, int tn, int N,
                                           v8f* acc) {
  long arow = g ? (long)g[rowm] : (long)rowm;
  const __bf16* Ap = A + arow * (long)K;
  const int koff = (lane >> 4) << 3;            // 0 or 8
  const int nt16 = N >> 4;
  const long fragBase = ((long)(tn >> 4) * 32 + lane) * 16;
  for (int k0 = 0; k0 < K; k0 += 32) {
    const __bf16* pa = Ap + k0 + koff;
    v8bf lo = *(const v8bf*)(pa);               // 16B load
    v8bf hi = *(const v8bf*)(pa + 16);          // 16B load
    v16bf a = __builtin_shufflevector(lo, hi,
              0,1,2,3,4,5,6,7,8,9,10,11,12,13,14,15);
    const __bf16* bb = Bp + (long)(k0 >> 5) * nt16 * (32 * 16) + fragBase;
#pragma unroll
    for (int j = 0; j < NT; ++j) {
      v16bf b = *(const v16bf*)(bb + (long)j * (32 * 16));  // 32B contiguous
      acc[j] = __builtin_amdgcn_wmma_f32_16x16x32_bf16(
                   false, a, false, b, (short)0, acc[j], false, false);
    }
  }
}

template <int ACT>   // 0=none 1=ssp 2=silu 3=sigmoid-gate
__global__ __launch_bounds__(256)
void gemm_kernel(GemmP p) {
  const int lane = threadIdx.x & 31;
  const int wave = threadIdx.x >> 5;
  const int tilesN = p.N >> 7;                  // 128-wide strips
  const long tid = (long)blockIdx.x * WAVES_PER_BLOCK + wave;
  const long totTiles = (long)(p.M >> 4) * tilesN;
  if (tid >= totTiles) return;
  const int tm = (int)(tid / tilesN);
  const int tn = (int)(tid % tilesN) << 7;
  const int m0 = tm << 4;
  const int rowm = m0 + (lane & 15);

  v8f acc[NT];
#pragma unroll
  for (int j = 0; j < NT; ++j)
#pragma unroll
    for (int r = 0; r < 8; ++r) acc[j][r] = 0.f;

  if (p.A0) gemm_accum(p.A0, p.g0, p.K0, p.B0, rowm, lane, tn, p.N, acc);
  if (p.A1) gemm_accum(p.A1, p.g1, p.K1, p.B1, rowm, lane, tn, p.N, acc);
  if (p.A2) gemm_accum(p.A2, p.g2, p.K2, p.B2, rowm, lane, tn, p.N, acc);

  // C layout: lanes 0-15 -> rows m0..m0+7, lanes 16-31 -> m0+8..m0+15
  const int mbase = m0 + ((lane >> 4) << 3);
  const int lcol = lane & 15;

  float bcol[NT], rcol[NT];
#pragma unroll
  for (int j = 0; j < NT; ++j) {
    const int col = tn + j * 16 + lcol;
    bcol[j] = p.bias ? p.bias[col] : 0.f;
    rcol[j] = p.r1n  ? p.r1n[col]  : 0.f;
  }

#pragma unroll
  for (int r = 0; r < 8; ++r) {
    const long m = mbase + r;
    const float r1 = p.r1m ? p.r1m[m] : 0.f;
    long scatRow = 0;
    if (p.scat) scatRow = (long)p.scat[m] * p.N;
#pragma unroll
    for (int j = 0; j < NT; ++j) {
      const int col = tn + j * 16 + lcol;
      float v = acc[j][r] + bcol[j];
      if (p.r1m) v = fmaf(r1, rcol[j], v);
      if (ACT == 1) v = act_ssp(v);
      if (ACT == 2) v = act_silu(v);
      const long idx = m * p.N + col;
      if (p.mulB) v *= p.mulB[idx];
      if (p.resB) v += p.resB[idx];
      if (ACT == 3) {
        float gte = 1.f / (1.f + __expf(-v));
        v = gte * p.gA[idx] + (1.f - gte) * p.gB[idx];
      }
      if (p.scat) {
        atomicAdd(p.Cf + scatRow + col, v);     // fused segment_sum
      } else if (p.Cf) {
        p.Cf[idx] = v;
      }
      if (p.Cb) p.Cb[idx] = (__bf16)v;
    }
  }
}

// ---------------------------------------------------------------------------
// Glue kernels
// ---------------------------------------------------------------------------
__global__ void k_f32_to_bf16(const float* __restrict__ s,
                              __bf16* __restrict__ d, long n) {
  long i = (long)blockIdx.x * blockDim.x + threadIdx.x;
  if (i < n) d[i] = (__bf16)s[i];
}

__global__ void k_node_embed(const float* __restrict__ x,
                             const float* __restrict__ nw,
                             const float* __restrict__ nb,
                             float* __restrict__ hsf, float* __restrict__ hef,
                             __bf16* __restrict__ hsb, __bf16* __restrict__ heb,
                             long total) {
  long i = (long)blockIdx.x * blockDim.x + threadIdx.x;
  if (i >= total) return;
  int j = (int)(i & 127);
  long n = i >> 7;
  float v = fmaf(x[n], nw[j], nb[j]);
  hsf[i] = v; hef[i] = v;
  hsb[i] = (__bf16)v; heb[i] = (__bf16)v;
}

__global__ void k_edge_prep(const float* __restrict__ ea,
                            const float* __restrict__ pos,
                            const int* __restrict__ s, const int* __restrict__ t,
                            const float* __restrict__ ew,
                            const float* __restrict__ eb,
                            __bf16* __restrict__ e3b, float* __restrict__ d2,
                            int E) {
  int e = blockIdx.x * blockDim.x + threadIdx.x;
  if (e >= E) return;
  float a0 = ea[e * 4 + 0], a1 = ea[e * 4 + 1];
  float a2 = ea[e * 4 + 2], a3 = ea[e * 4 + 3];
  int si = s[e], ti = t[e];
  float dx = pos[si * 3 + 0] - pos[ti * 3 + 0];
  float dy = pos[si * 3 + 1] - pos[ti * 3 + 1];
  float dz = pos[si * 3 + 2] - pos[ti * 3 + 2];
  float dd = dx * dx + dy * dy + dz * dz;
  float d = sqrtf(dd);
  d2[e] = dd;
  __bf16* out = e3b + (long)e * 288;
  for (int j = 0; j < 256; ++j) {
    float v = eb[j];
    v = fmaf(a0, ew[j], v);
    v = fmaf(a1, ew[256 + j], v);
    v = fmaf(a2, ew[512 + j], v);
    v = fmaf(a3, ew[768 + j], v);
    out[j] = (__bf16)v;
  }
  const float delta = 8.f / 31.f;
  const float coeff = -0.5f / (delta * delta);
  for (int j = 0; j < 32; ++j) {
    float u = d - (float)j * delta;
    out[256 + j] = (__bf16)__expf(coeff * u * u);
  }
}

// ---------------------------------------------------------------------------
// Host side
// ---------------------------------------------------------------------------
static inline void* ws_take(char*& p, size_t bytes) {
  void* r = (void*)p;
  p += (bytes + 255) & ~(size_t)255;
  return r;
}

static void gemm_go(const GemmP& p, int act, hipStream_t st) {
  long tiles = (long)(p.M / 16) * (p.N / 128);
  dim3 grid((unsigned)((tiles + WAVES_PER_BLOCK - 1) / WAVES_PER_BLOCK));
  dim3 blk(256);
  switch (act) {
    case 0: gemm_kernel<0><<<grid, blk, 0, st>>>(p); break;
    case 1: gemm_kernel<1><<<grid, blk, 0, st>>>(p); break;
    case 2: gemm_kernel<2><<<grid, blk, 0, st>>>(p); break;
    default: gemm_kernel<3><<<grid, blk, 0, st>>>(p); break;
  }
}

extern "C" void kernel_launch(void* const* d_in, const int* in_sizes, int n_in,
                              void* d_out, int out_size, void* d_ws, size_t ws_size,
                              hipStream_t stream) {
  (void)n_in; (void)out_size; (void)ws_size;
  const int Nn = in_sizes[0];       // 50000 (x is [N,1])
  const int E  = in_sizes[1] / 4;   // 800000
  const int L  = 4;

  const float* x         = (const float*)d_in[0];
  const float* edge_attr = (const float*)d_in[1];
  const float* pos       = (const float*)d_in[2];
  const int*   ei        = (const int*)  d_in[3];
  const float* node_w    = (const float*)d_in[4];
  const float* node_b    = (const float*)d_in[5];
  const float* edge_w    = (const float*)d_in[6];
  const float* edge_b    = (const float*)d_in[7];
  const float* sch_fw1   = (const float*)d_in[8];
  const float* sch_fb1   = (const float*)d_in[9];
  const float* sch_fw2   = (const float*)d_in[10];
  const float* sch_fb2   = (const float*)d_in[11];
  const float* sch_sw    = (const float*)d_in[12];
  const float* sch_sb    = (const float*)d_in[13];
  const float* sch_ow1   = (const float*)d_in[14];
  const float* sch_ob1   = (const float*)d_in[15];
  const float* sch_ow2   = (const float*)d_in[16];
  const float* sch_ob2   = (const float*)d_in[17];
  const float* eg_ew1    = (const float*)d_in[18];
  const float* eg_eb1    = (const float*)d_in[19];
  const float* eg_ew2    = (const float*)d_in[20];
  const float* eg_eb2    = (const float*)d_in[21];
  const float* eg_hw1    = (const float*)d_in[22];
  const float* eg_hb1    = (const float*)d_in[23];
  const float* eg_hw2    = (const float*)d_in[24];
  const float* eg_hb2    = (const float*)d_in[25];
  const float* gate_w    = (const float*)d_in[26];
  const float* gate_b    = (const float*)d_in[27];

  const int* sIdx = ei;
  const int* tIdx = ei + E;

  // ---- workspace layout -------------------------------------------------
  char* wp = (char*)d_ws;
  __bf16* e3b    = (__bf16*)ws_take(wp, (size_t)E * 288 * 2);
  float*  d2     = (float*) ws_take(wp, (size_t)E * 4);
  float*  hs_f   = (float*) ws_take(wp, (size_t)Nn * 128 * 4);
  float*  he_f   = (float*) ws_take(wp, (size_t)Nn * 128 * 4);
  __bf16* hs_b   = (__bf16*)ws_take(wp, (size_t)Nn * 128 * 2);
  __bf16* he_b   = (__bf16*)ws_take(wp, (size_t)Nn * 128 * 2);
  __bf16* T1b    = (__bf16*)ws_take(wp, (size_t)E * 256 * 2);   // edge intermediates
  float*  Ff     = (float*) ws_take(wp, (size_t)E * 128 * 4);   // SchNet filt
  float*  aggS_f = (float*) ws_take(wp, (size_t)Nn * 128 * 4);
  __bf16* aggS_b = (__bf16*)ws_take(wp, (size_t)Nn * 128 * 2);
  float*  aggE_f = (float*) ws_take(wp, (size_t)Nn * 256 * 4);
  __bf16* aggE_b = (__bf16*)ws_take(wp, (size_t)Nn * 256 * 2);
  __bf16* Ub     = (__bf16*)ws_take(wp, (size_t)Nn * 256 * 2);  // node intermediates

  // fused f32->bf16 conversion + WMMA-B-layout repack (per layer / per block)
  auto repack = [&](const float* src, int K, int N) -> __bf16* {
    size_t n = (size_t)K * N;
    __bf16* dst = (__bf16*)ws_take(wp, n * 2);
    int total = (K / 32) * (N / 16) * 32;
    k_repack_b<<<(total + 255) / 256, 256, 0, stream>>>(src, dst, K, N);
    return dst;
  };

  __bf16 *fw1p[4], *fw2p[4], *swp[4], *ow1p[4], *ow2p[4];
  __bf16 *ew1ap[4], *ew1bp[4], *ew1dp[4], *ew2p[4], *hw1ap[4], *hw1bp[4], *hw2p[4];
  for (int i = 0; i < L; ++i) {
    fw1p[i]  = repack(sch_fw1 + (size_t)i * 288 * 256, 288, 256);
    fw2p[i]  = repack(sch_fw2 + (size_t)i * 256 * 128, 256, 128);
    swp[i]   = repack(sch_sw  + (size_t)i * 128 * 128, 128, 128);
    ow1p[i]  = repack(sch_ow1 + (size_t)i * 128 * 128, 128, 128);
    ow2p[i]  = repack(sch_ow2 + (size_t)i * 128 * 128, 128, 128);
    const float* ew1 = eg_ew1 + (size_t)i * 545 * 256;
    ew1ap[i] = repack(ew1,             128, 256);   // rows 0..127   (he[s])
    ew1bp[i] = repack(ew1 + 128 * 256, 128, 256);   // rows 128..255 (he[t])
    ew1dp[i] = repack(ew1 + 257 * 256, 288, 256);   // rows 257..544 (e3)
    ew2p[i]  = repack(eg_ew2 + (size_t)i * 256 * 256, 256, 256);
    const float* hw1 = eg_hw1 + (size_t)i * 384 * 256;
    hw1ap[i] = repack(hw1,             128, 256);   // rows 0..127   (he)
    hw1bp[i] = repack(hw1 + 128 * 256, 256, 256);   // rows 128..383 (agg)
    hw2p[i]  = repack(eg_hw2 + (size_t)i * 256 * 128, 256, 128);
  }
  __bf16* gateap = repack(gate_w,             128, 128);  // rows 0..127   (hs)
  __bf16* gatebp = repack(gate_w + 128 * 128, 128, 128);  // rows 128..255 (he)

  // ---- embeddings -------------------------------------------------------
  {
    long tot = (long)Nn * 128;
    k_node_embed<<<(unsigned)((tot + 255) / 256), 256, 0, stream>>>(
        x, node_w, node_b, hs_f, he_f, hs_b, he_b, tot);
    k_edge_prep<<<(E + 255) / 256, 256, 0, stream>>>(
        edge_attr, pos, sIdx, tIdx, edge_w, edge_b, e3b, d2, E);
  }

  // ---- layers -----------------------------------------------------------
  for (int i = 0; i < L; ++i) {
    // ============ SchNet branch ============
    { // T1 = ssp(e3 @ fw1 + fb1)           [E,256] bf16
      GemmP p{}; p.A0 = e3b; p.K0 = 288; p.B0 = fw1p[i];
      p.bias = sch_fb1 + i * 256; p.Cb = T1b; p.M = E; p.N = 256;
      gemm_go(p, 1, stream);
    }
    { // filt = T1 @ fw2 + fb2              [E,128] f32
      GemmP p{}; p.A0 = T1b; p.K0 = 256; p.B0 = fw2p[i];
      p.bias = sch_fb2 + i * 128; p.Cf = Ff; p.M = E; p.N = 128;
      gemm_go(p, 0, stream);
    }
    hipMemsetAsync(aggS_f, 0, (size_t)Nn * 128 * 4, stream);
    { // agg[t] += (hs[s] @ sw + sb) * filt   (fused msg + segment_sum)
      GemmP p{}; p.A0 = hs_b; p.g0 = sIdx; p.K0 = 128; p.B0 = swp[i];
      p.bias = sch_sb + i * 128; p.mulB = Ff; p.scat = tIdx; p.Cf = aggS_f;
      p.M = E; p.N = 128;
      gemm_go(p, 0, stream);
    }
    {
      long tot = (long)Nn * 128;
      k_f32_to_bf16<<<(unsigned)((tot + 255) / 256), 256, 0, stream>>>(
          aggS_f, aggS_b, tot);
    }
    { // U = ssp(agg @ ow1 + ob1)           [Nn,128] bf16
      GemmP p{}; p.A0 = aggS_b; p.K0 = 128; p.B0 = ow1p[i];
      p.bias = sch_ob1 + i * 128; p.Cb = Ub; p.M = Nn; p.N = 128;
      gemm_go(p, 1, stream);
    }
    { // hs = hs + U @ ow2 + ob2
      GemmP p{}; p.A0 = Ub; p.K0 = 128; p.B0 = ow2p[i];
      p.bias = sch_ob2 + i * 128; p.resB = hs_f; p.Cf = hs_f; p.Cb = hs_b;
      p.M = Nn; p.N = 128;
      gemm_go(p, 0, stream);
    }

    // ============ EGNN branch ============
    { // M1 = silu(he[s]@Wa + he[t]@Wb + d2*wc + e3@Wd + eb1)  [E,256] bf16
      GemmP p{};
      p.A0 = he_b; p.g0 = sIdx; p.K0 = 128; p.B0 = ew1ap[i];
      p.A1 = he_b; p.g1 = tIdx; p.K1 = 128; p.B1 = ew1bp[i];
      p.A2 = e3b;               p.K2 = 288; p.B2 = ew1dp[i];
      p.r1m = d2; p.r1n = eg_ew1 + (size_t)i * 545 * 256 + 256 * 256; // row 256
      p.bias = eg_eb1 + i * 256; p.Cb = T1b; p.M = E; p.N = 256;
      gemm_go(p, 2, stream);
    }
    hipMemsetAsync(aggE_f, 0, (size_t)Nn * 256 * 4, stream);
    { // aggE[t] += silu(M1 @ ew2 + eb2)     (fused m + segment_sum)
      GemmP p{}; p.A0 = T1b; p.K0 = 256; p.B0 = ew2p[i];
      p.bias = eg_eb2 + i * 256; p.scat = tIdx; p.Cf = aggE_f;
      p.M = E; p.N = 256;
      gemm_go(p, 2, stream);
    }
    {
      long tot = (long)Nn * 256;
      k_f32_to_bf16<<<(unsigned)((tot + 255) / 256), 256, 0, stream>>>(
          aggE_f, aggE_b, tot);
    }
    { // U = silu(he@hw1a + aggE@hw1b + hb1)  [Nn,256] bf16
      GemmP p{};
      p.A0 = he_b;   p.K0 = 128; p.B0 = hw1ap[i];
      p.A1 = aggE_b; p.K1 = 256; p.B1 = hw1bp[i];
      p.bias = eg_hb1 + i * 256; p.Cb = Ub; p.M = Nn; p.N = 256;
      gemm_go(p, 2, stream);
    }
    { // he = he + U @ hw2 + hb2
      GemmP p{}; p.A0 = Ub; p.K0 = 256; p.B0 = hw2p[i];
      p.bias = eg_hb2 + i * 128; p.resB = he_f; p.Cf = he_f; p.Cb = he_b;
      p.M = Nn; p.N = 128;
      gemm_go(p, 0, stream);
    }
  }

  // ---- fusion gate: out = g*hs + (1-g)*he, g = sigmoid([hs|he]@gate_w+b)
  {
    GemmP p{};
    p.A0 = hs_b; p.K0 = 128; p.B0 = gateap;
    p.A1 = he_b; p.K1 = 128; p.B1 = gatebp;
    p.bias = gate_b; p.gA = hs_f; p.gB = he_f;
    p.Cf = (float*)d_out; p.M = Nn; p.N = 128;
    gemm_go(p, 3, stream);
  }
}
